// ObjectRelationNetwork_3882650436596
// MI455X (gfx1250) — compile-verified, matrix operand order checked by reading, not run on hardware
//
#include <hip/hip_runtime.h>

typedef __attribute__((ext_vector_type(16))) _Float16 v16h;
typedef __attribute__((ext_vector_type(8)))  _Float16 v8h;
typedef __attribute__((ext_vector_type(4)))  _Float16 v4h;
typedef __attribute__((ext_vector_type(8)))  float    v8f;
typedef __attribute__((ext_vector_type(4)))  float    v4f;
typedef int gvec4i __attribute__((vector_size(16)));   // matches builtin param type

#define NBT   240      // B * (T-1) = 16 * 15
#define KOBJ  32
#define DOBJ  128
#define HDIM  256

#define LDS_AS __attribute__((address_space(3)))
#define GLB_AS __attribute__((address_space(1)))

// Load a 16-lane-striped f16 fragment (A or B) from a row-major f16 matrix:
// per lane, elements 0..7 = [row][k0 + half*8 .. +7],
//           elements 8..15 = [row][k0 + 16 + half*8 .. +7].
__device__ __forceinline__ v16h load_frag16(const _Float16* base, int row, int ld,
                                            int k0, int half) {
    const _Float16* p = base + row * ld + k0 + half * 8;
    v8h lo = *(const v8h*)(p);
    v8h hi = *(const v8h*)(p + 16);
    v16h f;
#pragma unroll
    for (int i = 0; i < 8; ++i) { f[i] = lo[i]; f[i + 8] = hi[i]; }
    return f;
}

// B fragment from row-major f32 W1 (256 cols), converting to f16 on the fly.
__device__ __forceinline__ v16h load_bfrag_w1(const float* W1, int n, int kbase,
                                              int half) {
    const float* p = W1 + n * 256 + kbase + half * 8;
    v4f a = *(const v4f*)(p);
    v4f b = *(const v4f*)(p + 4);
    v4f c = *(const v4f*)(p + 16);
    v4f d = *(const v4f*)(p + 20);
    v16h f;
#pragma unroll
    for (int i = 0; i < 4; ++i) {
        f[i]      = (_Float16)a[i];
        f[i + 4]  = (_Float16)b[i];
        f[i + 8]  = (_Float16)c[i];
        f[i + 12] = (_Float16)d[i];
    }
    return f;
}

// ---------------------------------------------------------------------------
// W2 f32 -> f16 conversion (65536 elements).
__global__ __launch_bounds__(256) void kcvt_w2(const float* __restrict__ W2,
                                               _Float16* __restrict__ W2h) {
    int i = (blockIdx.x * 256 + threadIdx.x) * 4;
    v4f x = *(const v4f*)(W2 + i);
    v4h y;
#pragma unroll
    for (int q = 0; q < 4; ++q) y[q] = (_Float16)x[q];
    *(v4h*)(W2h + i) = y;
}

// ---------------------------------------------------------------------------
// Per (b,t): L = O_t @ W1[:, :128]^T, R = O_t @ W1[:, 128:]^T  (32 x 256 each),
// plus per-object row sums for the is_obj mask.
__global__ __launch_bounds__(256) void kprep(const float* __restrict__ O,
                                             const float* __restrict__ W1,
                                             float* __restrict__ Lws,
                                             float* __restrict__ Rws,
                                             float* __restrict__ sws) {
    int bt = blockIdx.x;
    int b = bt / 15, t = bt % 15;
    const float* Ot = O + ((size_t)(b * 16 + t + 1)) * KOBJ * DOBJ;

    __shared__ _Float16 sO[KOBJ * DOBJ];   // 8 KB
    int tid = threadIdx.x;

    {
        const float* src = Ot + tid * 16;
        _Float16*    dst = sO + tid * 16;
#pragma unroll
        for (int v = 0; v < 4; ++v) {
            v4f x = *(const v4f*)(src + v * 4);
            v4h y;
#pragma unroll
            for (int q = 0; q < 4; ++q) y[q] = (_Float16)x[q];
            *(v4h*)(dst + v * 4) = y;
        }
    }
    if (tid < KOBJ) {
        const float* r = Ot + tid * DOBJ;
        float sum = 0.f;
#pragma unroll 4
        for (int c = 0; c < DOBJ; c += 4) {
            v4f x = *(const v4f*)(r + c);
            sum += x[0] + x[1] + x[2] + x[3];
        }
        sws[bt * KOBJ + tid] = sum;
    }
    __syncthreads();

    int wave = tid >> 5, lane = tid & 31, half = lane >> 4;
    int m0   = (wave & 1) * 16;
    int row  = m0 + (lane & 15);
    int nbase = (wave >> 1) * 64;
    int ncol  = lane & 15;

    v8f accL[4] = {};
    v8f accR[4] = {};
#pragma unroll
    for (int k0 = 0; k0 < 128; k0 += 32) {
        v16h a = load_frag16(sO, row, DOBJ, k0, half);
#pragma unroll
        for (int nt = 0; nt < 4; ++nt) {
            int n = nbase + nt * 16 + ncol;
            v16h bL = load_bfrag_w1(W1, n, k0, half);
            accL[nt] = __builtin_amdgcn_wmma_f32_16x16x32_f16(
                false, a, false, bL, (short)0, accL[nt], false, false);
            v16h bR = load_bfrag_w1(W1, n, 128 + k0, half);
            accR[nt] = __builtin_amdgcn_wmma_f32_16x16x32_f16(
                false, a, false, bR, (short)0, accR[nt], false, false);
        }
    }

    float* Lb = Lws + (size_t)bt * KOBJ * HDIM;
    float* Rb = Rws + (size_t)bt * KOBJ * HDIM;
#pragma unroll
    for (int nt = 0; nt < 4; ++nt) {
        int n = nbase + nt * 16 + ncol;
#pragma unroll
        for (int r = 0; r < 8; ++r) {
            int rr = m0 + half * 8 + r;
            Lb[rr * HDIM + n] = accL[nt][r];
            Rb[rr * HDIM + n] = accR[nt][r];
        }
    }
}

// ---------------------------------------------------------------------------
// Main kernel: stage W2 (f16, 128 KB) into LDS via async DMA while building
// h = relu(L_i + R_j + b1) (128x256 f16, 64 KB LDS), then e = relu(h @ W2^T + b2).
// Wave tiling: 4 M-tiles x 4 N-tiles per wave (A/B frags reused 4x each).
__global__ __launch_bounds__(256) void kmain(const float* __restrict__ Lws,
                                             const float* __restrict__ Rws,
                                             const float* __restrict__ b1,
                                             const _Float16* __restrict__ W2h,
                                             const float* __restrict__ b2,
                                             float* __restrict__ e) {
    int bt = blockIdx.x >> 3;
    int mb = blockIdx.x & 7;

    __shared__ _Float16 sH[128 * HDIM];    // 64 KB
    __shared__ _Float16 sW[HDIM * HDIM];   // 128 KB
    int tid = threadIdx.x;

    // --- Kick off async W2h -> LDS staging (32 x b128 per thread) -----------
#if __has_builtin(__builtin_amdgcn_global_load_async_to_lds_b128)
#pragma unroll 4
    for (int c = 0; c < 32; ++c) {
        int idx = (c * 256 + tid) * 8;     // 8 halves = 16 bytes
        __builtin_amdgcn_global_load_async_to_lds_b128(
            (GLB_AS gvec4i*)(W2h + idx),
            (LDS_AS gvec4i*)(sW + idx), 0, 0);
    }
#else
#pragma unroll 4
    for (int c = 0; c < 32; ++c) {
        int idx = (c * 256 + tid) * 8;
        *(v8h*)(sW + idx) = *(const v8h*)(W2h + idx);
    }
#endif

    // --- Build h tile while the DMA runs ------------------------------------
    {
        int rowl = tid >> 1;
        int colb = (tid & 1) * 128;
        int p = mb * 128 + rowl;
        int i = p >> 5, j = p & 31;
        const float* Lp  = Lws + ((size_t)bt * KOBJ + i) * HDIM + colb;
        const float* Rp  = Rws + ((size_t)bt * KOBJ + j) * HDIM + colb;
        const float* b1p = b1 + colb;
        _Float16*    hp  = sH + rowl * HDIM + colb;
#pragma unroll 4
        for (int c = 0; c < 128; c += 4) {
            v4f l  = *(const v4f*)(Lp + c);
            v4f r  = *(const v4f*)(Rp + c);
            v4f bb = *(const v4f*)(b1p + c);
            v4h y;
#pragma unroll
            for (int q = 0; q < 4; ++q) {
                float v = l[q] + r[q] + bb[q];
                v = v > 0.f ? v : 0.f;
                y[q] = (_Float16)v;
            }
            *(v4h*)(hp + c) = y;
        }
    }

#if __has_builtin(__builtin_amdgcn_s_wait_asynccnt)
    __builtin_amdgcn_s_wait_asynccnt(0);
#elif __has_builtin(__builtin_amdgcn_global_load_async_to_lds_b128)
    asm volatile("s_wait_asynccnt 0" ::: "memory");
#endif
    __syncthreads();

    // --- GEMM: 8 waves = 2 M-groups x 4 N-groups, 4x4 tiles per wave --------
    int wave = tid >> 5, lane = tid & 31, half = lane >> 4;
    int mg = wave >> 2;                    // 0..1
    int ng = wave & 3;                     // 0..3
    int mrow0 = mg * 64;
    int mlane = lane & 15;
    int ncol  = lane & 15;

    v8f acc[4][4] = {};
#pragma unroll
    for (int k0 = 0; k0 < 256; k0 += 32) {
        v16h a[4], bf[4];
#pragma unroll
        for (int mt = 0; mt < 4; ++mt)
            a[mt] = load_frag16(sH, mrow0 + mt * 16 + mlane, HDIM, k0, half);
#pragma unroll
        for (int nt = 0; nt < 4; ++nt)
            bf[nt] = load_frag16(sW, (ng * 4 + nt) * 16 + ncol, HDIM, k0, half);
#pragma unroll
        for (int mt = 0; mt < 4; ++mt)
#pragma unroll
            for (int nt = 0; nt < 4; ++nt)
                acc[mt][nt] = __builtin_amdgcn_wmma_f32_16x16x32_f16(
                    false, a[mt], false, bf[nt], (short)0, acc[mt][nt], false, false);
    }

    // --- Epilogue: bias + relu, coalesced b32 stores ------------------------
    float bias[4];
#pragma unroll
    for (int nt = 0; nt < 4; ++nt) bias[nt] = b2[(ng * 4 + nt) * 16 + ncol];

#pragma unroll
    for (int mt = 0; mt < 4; ++mt) {
        size_t rowbase = (size_t)bt * 1024 + mb * 128 + mrow0 + mt * 16 + half * 8;
#pragma unroll
        for (int nt = 0; nt < 4; ++nt) {
            int n = (ng * 4 + nt) * 16 + ncol;
#pragma unroll
            for (int r = 0; r < 8; ++r) {
                float v = acc[mt][nt][r] + bias[nt];
                v = v > 0.f ? v : 0.f;
                e[(rowbase + r) * HDIM + n] = v;
            }
        }
    }
}

// ---------------------------------------------------------------------------
// all_is_obj = clip(clip(sum_i,0,1)*clip(sum_j,0,1)*(t+1), 0, 1)
__global__ __launch_bounds__(256) void kobj(const float* __restrict__ sws,
                                            float* __restrict__ out2) {
    int bt = blockIdx.x;
    float cnt = (float)((bt % 15) + 1);
    __shared__ float c[KOBJ];
    int tid = threadIdx.x;
    if (tid < KOBJ) {
        float v = sws[bt * KOBJ + tid];
        v = v < 0.f ? 0.f : (v > 1.f ? 1.f : v);
        c[tid] = v;
    }
    __syncthreads();
#pragma unroll
    for (int q = 0; q < 4; ++q) {
        int p = tid * 4 + q;
        int i = p >> 5, j = p & 31;
        float v = c[i] * c[j] * cnt;
        v = v > 1.f ? 1.f : v;
        out2[(size_t)bt * 1024 + p] = v;
    }
}

extern "C" void kernel_launch(void* const* d_in, const int* in_sizes, int n_in,
                              void* d_out, int out_size, void* d_ws, size_t ws_size,
                              hipStream_t stream) {
    const float* O  = (const float*)d_in[0];   // (16,16,32,128)
    const float* W1 = (const float*)d_in[1];   // (256,256)
    const float* b1 = (const float*)d_in[2];   // (256,)
    const float* W2 = (const float*)d_in[3];   // (256,256)
    const float* b2 = (const float*)d_in[4];   // (256,)

    char* ws = (char*)d_ws;
    const size_t LR_BYTES = (size_t)NBT * KOBJ * HDIM * sizeof(float); // 7.86 MB
    float*     Lws = (float*)(ws);
    float*     Rws = (float*)(ws + LR_BYTES);
    _Float16*  W2h = (_Float16*)(ws + 2 * LR_BYTES);
    float*     sws = (float*)(ws + 2 * LR_BYTES + (size_t)HDIM * HDIM * sizeof(_Float16));

    float* e    = (float*)d_out;                       // 240*1024*256
    float* out2 = e + (size_t)NBT * 1024 * HDIM;       // 240*1024

    kcvt_w2<<<64, 256, 0, stream>>>(W2, W2h);
    kprep  <<<NBT, 256, 0, stream>>>(O, W1, Lws, Rws, sws);
    kmain  <<<NBT * 8, 256, 0, stream>>>(Lws, Rws, b1, W2h, b2, e);
    kobj   <<<NBT, 256, 0, stream>>>(sws, out2);
}
